// FFNMoE_60370060312670
// MI455X (gfx1250) — matrix operand order repaired and checked by reading, not compiled, not run on hardware
//
#include <hip/hip_runtime.h>

// Problem constants (match reference).
#define T_TOK 8192
#define D_DIM 1024
#define F_DIM 4096
#define E_EXP 8
#define S_SEQ 2048

// Workspace requirement: ~136 MB
//   probs   [T,8] f32, combine [T,8] f32, list [8,T] i32, counts[8],
//   impSum[8], loadSum[8], hbuf [T,F] f32 (reused per expert).

typedef __attribute__((ext_vector_type(16))) __bf16 bf16x16;
typedef __attribute__((ext_vector_type(8)))  __bf16 bf16x8;
typedef __attribute__((ext_vector_type(4)))  __bf16 bf16x4;
typedef __attribute__((ext_vector_type(8)))  float  f32x8;
typedef __attribute__((ext_vector_type(4)))  float  f32x4;

__device__ __forceinline__ bf16x16 cat16(bf16x8 a, bf16x8 b) {
  bf16x16 r;
#pragma unroll
  for (int i = 0; i < 8; ++i) { r[i] = a[i]; r[i + 8] = b[i]; }
  return r;
}

__device__ __forceinline__ f32x8 wmma_bf16(bf16x16 a, bf16x16 b, f32x8 c) {
  return __builtin_amdgcn_wmma_f32_16x16x32_bf16(false, a, false, b, (short)0, c,
                                                 false, false);
}

__device__ __forceinline__ void atom_add_f32(float* p, float v) {
  unsafeAtomicAdd(p, v);   // lowers to global_atomic_add_f32
}

// ---------------------------------------------------------------- init
__global__ __launch_bounds__(256) void k_init(float* out, int* list, int* counts) {
  size_t gid = (size_t)blockIdx.x * 256 + threadIdx.x;
  size_t nthreads = (size_t)gridDim.x * 256;
  size_t N = (size_t)T_TOK * D_DIM + 1;
  for (size_t j = gid; j < N; j += nthreads) out[j] = 0.0f;
  if (gid < (size_t)E_EXP * T_TOK) list[gid] = -1;
  if (gid < E_EXP) counts[gid] = 0;
}

// ---------------------------------------------------------------- routing
__global__ __launch_bounds__(256) void k_routing(
    const float* __restrict__ tok, const float* __restrict__ cond,
    const float* __restrict__ Wrt, const float* __restrict__ brt,
    const float* __restrict__ Wrc,
    float* __restrict__ probsOut, float* __restrict__ combineOut,
    int* __restrict__ list, int* __restrict__ counts) {
  int t = blockIdx.x;
  int b = t / S_SEQ;
  int tid = threadIdx.x;
  int e = tid & 7, dbase = tid >> 3;       // 8 experts x 32 d-lanes
  const float* tr = tok + (size_t)t * D_DIM;
  const float* cr = cond + (size_t)b * D_DIM;
  float p = 0.0f;
#pragma unroll 4
  for (int j = 0; j < D_DIM / 32; ++j) {
    int d = dbase + j * 32;
    p += tr[d] * Wrt[d * E_EXP + e] + cr[d] * Wrc[d * E_EXP + e];
  }
  __shared__ float sred[256];
  sred[tid] = p;
  __syncthreads();
  for (int s = 128; s >= 8; s >>= 1) {
    if (tid < s) sred[tid] += sred[tid + s];
    __syncthreads();
  }
  if (tid == 0) {
    float l[E_EXP], pr[E_EXP];
    float mx = -1e30f;
#pragma unroll
    for (int i = 0; i < E_EXP; ++i) { l[i] = sred[i] + brt[i]; mx = fmaxf(mx, l[i]); }
    float sum = 0.0f;
#pragma unroll
    for (int i = 0; i < E_EXP; ++i) { pr[i] = __expf(l[i] - mx); sum += pr[i]; }
    float inv = 1.0f / sum;
#pragma unroll
    for (int i = 0; i < E_EXP; ++i) {
      pr[i] *= inv;
      probsOut[(size_t)t * E_EXP + i] = pr[i];
      combineOut[(size_t)t * E_EXP + i] = 0.0f;
    }
    // top-2 (lowest index wins ties, like lax.top_k)
    int i0 = 0;
#pragma unroll
    for (int i = 1; i < E_EXP; ++i) if (pr[i] > pr[i0]) i0 = i;
    int i1 = (i0 == 0) ? 1 : 0;
#pragma unroll
    for (int i = 0; i < E_EXP; ++i)
      if (i != i0 && i != i1 && pr[i] > pr[i1]) i1 = i;
    float s2 = pr[i0] + pr[i1];
    float inv2 = 1.0f / fmaxf(s2, 1e-6f);
    combineOut[(size_t)t * E_EXP + i0] = pr[i0] * inv2;
    combineOut[(size_t)t * E_EXP + i1] = pr[i1] * inv2;
    int p0 = atomicAdd(&counts[i0], 1);
    list[i0 * T_TOK + p0] = t;
    int p1 = atomicAdd(&counts[i1], 1);
    list[i1 * T_TOK + p1] = t;
  }
}

// ---------------------------------------------------------------- GEMM
// MODE 0: h = gelu(gather(tokens) @ W1[e] + b1[e])        (Kdim=D, Ndim=F)
// MODE 1: out[tok] += combine[tok,e]*(h @ W2[e] + b2[e])  (Kdim=F, Ndim=D)
// Tile BM=BN=128, BK=32; 8 waves, wave = 64x32 (4x2 frags of 16x16).
// f32 operands split on the fly into bf16 hi/lo; 3 WMMAs per frag pair.
// Software pipeline: double-buffered LDS (1 barrier/iter) + register
// prefetch of the next k-tile's global loads before the WMMA burst.
#define BKP 40   // padded LDS K-stride (bf16 elems) to spread banks

template <int MODE>
__global__ __launch_bounds__(256) void moe_gemm(
    const float* __restrict__ Abase,   // tokens [T,D] or hbuf [.,F]
    const float* __restrict__ Bbase,   // W1[e] [D,F] or W2[e] [F,D]
    const float* __restrict__ bias,    // b1[e] [F]   or b2[e] [D]
    const int* __restrict__ list,      // token list for expert e
    const int* __restrict__ cntPtr,    // &counts[e]
    const float* __restrict__ combine, // [T,8] (MODE 1)
    float* __restrict__ Obase,         // hbuf or out
    int Kdim, int Ndim, int expertE) {
  int cnt = *cntPtr;
  int mtiles = (cnt + 127) >> 7;
  if ((int)blockIdx.y >= mtiles) return;
  const int row0 = blockIdx.y * 128;
  const int n0 = blockIdx.x * 128;

  __shared__ __bf16 sAh[2][128][BKP];
  __shared__ __bf16 sAl[2][128][BKP];
  __shared__ __bf16 sBh[2][128][BKP];   // transposed: [n][k]
  __shared__ __bf16 sBl[2][128][BKP];

  const int tid = threadIdx.x;
  const int lane = tid & 31;
  const int wid = tid >> 5;
  const int wm = wid >> 2;   // 0..1
  const int wn = wid & 3;    // 0..3

  // Per-thread A-staging sources (4 float4 quads per thread).
  const float* aptr[4];
#pragma unroll
  for (int i = 0; i < 4; ++i) {
    int q = tid + 256 * i;
    int r = q >> 3;            // row in tile
    int c = (q & 7) * 4;       // col in k-chunk
    if (MODE == 0) {
      int tokidx = list[row0 + r];
      aptr[i] = (tokidx >= 0) ? (Abase + (size_t)tokidx * Kdim + c) : nullptr;
    } else {
      aptr[i] = Abase + (size_t)(row0 + r) * Kdim + c;
    }
  }

  // Issue global loads for k-tile at offset k0 into registers.
  auto loadTiles = [&](int k0, f32x4 va[4], f32x4 vb[4]) {
#pragma unroll
    for (int i = 0; i < 4; ++i) {
      if (MODE == 0) {
#pragma unroll
        for (int j = 0; j < 4; ++j) va[i][j] = 0.0f;
        if (aptr[i]) va[i] = *(const f32x4*)(aptr[i] + k0);
      } else {
        va[i] = *(const f32x4*)(aptr[i] + k0);
      }
    }
#pragma unroll
    for (int i = 0; i < 4; ++i) {
      int q = tid + 256 * i;
      int k = q >> 5;            // 0..31
      int n = (q & 31) * 4;      // 0..124
      vb[i] = *(const f32x4*)(Bbase + (size_t)(k0 + k) * Ndim + n0 + n);
    }
  };

  // Convert hi/lo bf16 and store into LDS buffer `buf`.
  auto stageTiles = [&](int buf, const f32x4 va[4], const f32x4 vb[4]) {
#pragma unroll
    for (int i = 0; i < 4; ++i) {
      int q = tid + 256 * i;
      int r = q >> 3;
      int c = (q & 7) * 4;
      bf16x4 hv, lv;
#pragma unroll
      for (int j = 0; j < 4; ++j) {
        __bf16 h = (__bf16)va[i][j];
        hv[j] = h;
        lv[j] = (__bf16)(va[i][j] - (float)h);
      }
      *(bf16x4*)&sAh[buf][r][c] = hv;
      *(bf16x4*)&sAl[buf][r][c] = lv;
    }
#pragma unroll
    for (int i = 0; i < 4; ++i) {
      int q = tid + 256 * i;
      int k = q >> 5;
      int n = (q & 31) * 4;
#pragma unroll
      for (int j = 0; j < 4; ++j) {
        __bf16 h = (__bf16)vb[i][j];
        sBh[buf][n + j][k] = h;
        sBl[buf][n + j][k] = (__bf16)(vb[i][j] - (float)h);
      }
    }
  };

  f32x8 zero8;
#pragma unroll
  for (int i = 0; i < 8; ++i) zero8[i] = 0.0f;
  f32x8 acc[4][2];
#pragma unroll
  for (int mi = 0; mi < 4; ++mi)
#pragma unroll
    for (int ni = 0; ni < 2; ++ni) acc[mi][ni] = zero8;

  const int mrow = lane & 15;
  const int kb = (lane >> 4) * 8;     // A frag: K-base 0 or 8
  const int koff = (lane >> 4) * 16;  // unused; B frag also uses kb (0 or 8)

  f32x4 va[4], vb[4];
  loadTiles(0, va, vb);
  stageTiles(0, va, vb);
  __syncthreads();

  const int KT = Kdim / 32;
  for (int kt = 0; kt < KT; ++kt) {
    const int buf = kt & 1;
    const bool more = (kt + 1) < KT;
    // Prefetch next tile's global data before the WMMA burst.
    if (more) loadTiles((kt + 1) * 32, va, vb);

    // ---- fragments (documented wave32 layouts) + WMMA
    bf16x16 ah[4], al[4], bh[2], bl[2];
#pragma unroll
    for (int mi = 0; mi < 4; ++mi) {
      int m = wm * 64 + mi * 16 + mrow;
      ah[mi] = cat16(*(const bf16x8*)&sAh[buf][m][kb],
                     *(const bf16x8*)&sAh[buf][m][kb + 16]);
      al[mi] = cat16(*(const bf16x8*)&sAl[buf][m][kb],
                     *(const bf16x8*)&sAl[buf][m][kb + 16]);
    }
#pragma unroll
    for (int ni = 0; ni < 2; ++ni) {
      int n = wn * 32 + ni * 16 + mrow;
      bh[ni] = cat16(*(const bf16x8*)&sBh[buf][n][kb],
                     *(const bf16x8*)&sBh[buf][n][kb + 16]);
      bl[ni] = cat16(*(const bf16x8*)&sBl[buf][n][kb],
                     *(const bf16x8*)&sBl[buf][n][kb + 16]);
    }
#pragma unroll
    for (int mi = 0; mi < 4; ++mi)
#pragma unroll
      for (int ni = 0; ni < 2; ++ni) {
        acc[mi][ni] = wmma_bf16(ah[mi], bh[ni], acc[mi][ni]);
        acc[mi][ni] = wmma_bf16(ah[mi], bl[ni], acc[mi][ni]);
        acc[mi][ni] = wmma_bf16(al[mi], bh[ni], acc[mi][ni]);
      }

    // Stage the prefetched tile into the other LDS buffer.
    if (more) stageTiles(buf ^ 1, va, vb);
    __syncthreads();
  }

  // ---- epilogue (C frag: lane -> M = v + 8*(lane>=16), N = lane&15)
  const int hi8 = (lane >> 4) * 8;
#pragma unroll
  for (int mi = 0; mi < 4; ++mi) {
#pragma unroll
    for (int ni = 0; ni < 2; ++ni) {
      int n = n0 + wn * 32 + ni * 16 + mrow;
      float bn = bias[n];
      f32x8 a = acc[mi][ni];
#pragma unroll
      for (int v = 0; v < 8; ++v) {
        int mloc = wm * 64 + mi * 16 + hi8 + v;
        int slot = row0 + mloc;
        if (MODE == 0) {
          float x = a[v] + bn;
          float g = 0.5f * x * (1.0f + erff(x * 0.70710678118654752f));
          Obase[(size_t)slot * Ndim + n] = g;
        } else {
          int tokidx = list[slot];
          if (tokidx >= 0) {
            float w = combine[(size_t)tokidx * E_EXP + expertE];
            atom_add_f32(&Obase[(size_t)tokidx * Ndim + n], w * (a[v] + bn));
          }
        }
      }
    }
  }
}

// ------------------------------------------------------- aux reductions
__global__ __launch_bounds__(256) void k_reduce_aux(
    const float* __restrict__ probs, const float* __restrict__ combine,
    float* __restrict__ impSum, float* __restrict__ loadSum) {
  int e = blockIdx.x;
  int tid = threadIdx.x;
  float si = 0.0f, sl = 0.0f;
  for (int t = tid; t < T_TOK; t += 256) {
    si += probs[(size_t)t * E_EXP + e];
    sl += combine[(size_t)t * E_EXP + e];
  }
  __shared__ float sa[256], sb[256];
  sa[tid] = si; sb[tid] = sl;
  __syncthreads();
  for (int s = 128; s > 0; s >>= 1) {
    if (tid < s) { sa[tid] += sa[tid + s]; sb[tid] += sb[tid + s]; }
    __syncthreads();
  }
  if (tid == 0) { impSum[e] = sa[0]; loadSum[e] = sb[0]; }
}

__global__ void k_finalize(const float* __restrict__ impSum,
                           const float* __restrict__ loadSum,
                           float* __restrict__ out) {
  if (threadIdx.x == 0 && blockIdx.x == 0) {
    float imp[E_EXP], ld[E_EXP];
    float st = 0.0f, lt = 0.0f;
#pragma unroll
    for (int e = 0; e < E_EXP; ++e) {
      imp[e] = impSum[e] / (float)T_TOK;
      st += imp[e];
      ld[e] = loadSum[e];
      lt += ld[e];
    }
    st = fmaxf(st, 1e-6f);
    lt = fmaxf(lt, 1e-6f);
    float a = 0.0f, b = 0.0f;
    const float u = 1.0f / E_EXP;
#pragma unroll
    for (int e = 0; e < E_EXP; ++e) {
      float di = imp[e] / st - u;
      float dl = ld[e] / lt - u;
      a += di * di;
      b += dl * dl;
    }
    out[(size_t)T_TOK * D_DIM] = a / E_EXP + b / E_EXP;
  }
}

// ---------------------------------------------------------------- launch
extern "C" void kernel_launch(void* const* d_in, const int* in_sizes, int n_in,
                              void* d_out, int out_size, void* d_ws, size_t ws_size,
                              hipStream_t stream) {
  const float* tokenHidden = (const float*)d_in[0];
  const float* conditionHidden = (const float*)d_in[1];
  const float* W1 = (const float*)d_in[2];
  const float* b1 = (const float*)d_in[3];
  const float* W2 = (const float*)d_in[4];
  const float* b2 = (const float*)d_in[5];
  const float* Wrt = (const float*)d_in[6];
  const float* brt = (const float*)d_in[7];
  const float* Wrc = (const float*)d_in[8];
  float* out = (float*)d_out;

  char* w = (char*)d_ws;
  float* probs   = (float*)(w);                         // T*8 f32
  float* combine = (float*)(w + 262144);                // T*8 f32
  int*   list    = (int*)  (w + 524288);                // 8*T i32
  int*   counts  = (int*)  (w + 786432);                // 8 i32
  float* impSum  = (float*)(w + 786688);                // 8 f32
  float* loadSum = (float*)(w + 786944);                // 8 f32
  float* hbuf    = (float*)(w + 1048576);               // T*F f32 (134 MB)

  k_init<<<8192, 256, 0, stream>>>(out, list, counts);
  k_routing<<<T_TOK, 256, 0, stream>>>(tokenHidden, conditionHidden, Wrt, brt, Wrc,
                                       probs, combine, list, counts);
  k_reduce_aux<<<E_EXP, 256, 0, stream>>>(probs, combine, impSum, loadSum);

  for (int e = 0; e < E_EXP; ++e) {
    const float* W1e = W1 + (size_t)e * D_DIM * F_DIM;
    const float* b1e = b1 + (size_t)e * F_DIM;
    const float* W2e = W2 + (size_t)e * F_DIM * D_DIM;
    const float* b2e = b2 + (size_t)e * D_DIM;
    // GEMM1: [cnt_e, D] x [D, F] -> gelu -> hbuf
    moe_gemm<0><<<dim3(F_DIM / 128, T_TOK / 128), 256, 0, stream>>>(
        tokenHidden, W1e, b1e, list + e * T_TOK, counts + e, nullptr, hbuf,
        D_DIM, F_DIM, e);
    // GEMM2: [cnt_e, F] x [F, D] -> weighted atomic add into out
    moe_gemm<1><<<dim3(D_DIM / 128, T_TOK / 128), 256, 0, stream>>>(
        hbuf, W2e, b2e, list + e * T_TOK, counts + e, combine, out,
        F_DIM, D_DIM, e);
  }

  k_finalize<<<1, 32, 0, stream>>>(impSum, loadSum, out);
}
